// ChaPromptGenBlock_36009005809798
// MI455X (gfx1250) — matrix operand order repaired and checked by reading, not compiled, not run on hardware
//
#include <hip/hip_runtime.h>
#include <hip/hip_bf16.h>
#include <math.h>

typedef __attribute__((ext_vector_type(16))) __bf16 v16bf;
typedef __attribute__((ext_vector_type(8)))  __bf16 v8bf;
typedef __attribute__((ext_vector_type(8)))  float  v8f;

#define DIMC 128
#define HIDDEN 340
#define HPAD 352
#define NEXP 4
#define HW 36864              // 192*192
#define TOK_PER_WG 128
#define WGS_PER_IMG 288       // 36864/128
#define NWG 576               // 2*288
#define PITCH_X 129           // f32 pitch (stride%64==1 -> conflict free)
#define PITCH_K 136           // bf16 pitch for K=128 buffers
#define PITCH_H 360           // bf16 pitch for K=352 buffers

// ---- workspace byte offsets ----
#define OFF_EMB    0
#define OFF_SCALE  2048
#define OFF_STATS  4096       // imp[4], load[4]
#define OFF_WC     8192       // 128*136 bf16
#define OFF_W1     43008      // 4*352*136 bf16   (Bt1[e][h][c])
#define OFF_W2     425984     // 4*128*360 bf16   (Bt2[e][o][h])
#define N_W1 (4*HPAD*PITCH_K) // 191488 elems
#define N_W2 (4*128*PITCH_H)  // 184320 elems

// ---- LDS layout (bytes) ----
#define L_SX   0              // 128*129*4 = 66048   raw x tokens f32
#define L_SSF  66048          // 128*136*2 = 34816   sf bf16 (conv A)
#define L_STY  100864         // 128*136*2 = 34816   ty bf16 (fc1 A)
#define L_SH   135680         // 128*360*2 = 92160   h bf16 (fc2 A); reused as y f32
#define L_SW   227840         // 352*136*2 = 95744   staged weights (max)
#define L_SG   323584         // 128*4*4   = 2048    gates (aliased with w_gate stage)
#define L_SSC  325632         // 128*4     = 512     scale row
#define SMEM_MAIN 326144      // <= 327680 (320KB)

__device__ __forceinline__ v16bf mk16(v8bf lo, v8bf hi) {
  v16bf r;
#pragma unroll
  for (int j = 0; j < 8; ++j) { r[j] = lo[j]; r[8 + j] = hi[j]; }
  return r;
}

__device__ __forceinline__ v16bf ldfragA(const __bf16* row, int k0, int off8) {
  return mk16(*(const v8bf*)(row + k0 + off8), *(const v8bf*)(row + k0 + 16 + off8));
}

__device__ __forceinline__ v16bf ldfragB(const __bf16* row, int k0, int off16) {
  return mk16(*(const v8bf*)(row + k0 + off16 / 2), *(const v8bf*)(row + k0 + 16 + off16 / 2));
}

__device__ __forceinline__ v8f wmma_bf16(v16bf a, v16bf b, v8f c) {
  return __builtin_amdgcn_wmma_f32_16x16x32_bf16(false, a, false, b, (short)0, c, false, false);
}

// ---------------- emb = mean over H,W ----------------
__global__ void k_emb(const float* __restrict__ x, float* __restrict__ emb) {
  __shared__ float red[256];
  const int b = blockIdx.x >> 7, c = blockIdx.x & 127;
  const float* p = x + ((size_t)b * 128 + c) * HW;
  float s = 0.f;
  for (int i = threadIdx.x; i < HW; i += 256) s += p[i];
  red[threadIdx.x] = s;
  __syncthreads();
  for (int st = 128; st > 0; st >>= 1) {
    if (threadIdx.x < st) red[threadIdx.x] += red[threadIdx.x + st];
    __syncthreads();
  }
  if (threadIdx.x == 0) emb[blockIdx.x] = red[0] * (1.f / (float)HW);
}

// ---------------- prompt softmax -> per-channel scale; zero stats ----------------
__global__ void k_spectral(const float* __restrict__ prompt, const float* __restrict__ wlin,
                           const float* __restrict__ blin, char* __restrict__ ws) {
  float* emb = (float*)(ws + OFF_EMB);
  float* scale = (float*)(ws + OFF_SCALE);
  float* stats = (float*)(ws + OFF_STATS);
  __shared__ float sl[10], spw[10];
  const int t = threadIdx.x;
  if (t < 10) {
    int b = t / 5, p = t % 5;
    float a = blin[p];
    for (int c = 0; c < 128; ++c) a += emb[b * 128 + c] * wlin[p * 128 + c];
    sl[t] = a;
  }
  if (t >= 32 && t < 40) stats[t - 32] = 0.f;  // zero importance/load accumulators
  __syncthreads();
  if (t < 2) {
    float m = sl[t * 5];
    for (int p = 1; p < 5; ++p) m = fmaxf(m, sl[t * 5 + p]);
    float e[5], s = 0.f;
    for (int p = 0; p < 5; ++p) { e[p] = expf(sl[t * 5 + p] - m); s += e[p]; }
    for (int p = 0; p < 5; ++p) spw[t * 5 + p] = e[p] / s;
  }
  __syncthreads();
  for (int idx = t; idx < 256; idx += 64) {
    int b = idx >> 7, c = idx & 127;
    float s = 0.f;
    for (int p = 0; p < 5; ++p) s += spw[b * 5 + p] * prompt[p * 128 + c];
    scale[idx] = s;
  }
}

// ---------------- weights -> bf16, transposed B-layout, hidden padded to 352 ----------------
__global__ void k_prep(const float* __restrict__ wconv, const float* __restrict__ w1,
                       const float* __restrict__ w2, char* __restrict__ ws) {
  __bf16* wc = (__bf16*)(ws + OFF_WC);
  __bf16* wb1 = (__bf16*)(ws + OFF_W1);
  __bf16* wb2 = (__bf16*)(ws + OFF_W2);
  const int NC = 128 * 128;
  int idx = blockIdx.x * 256 + threadIdx.x;
  if (idx < NC) {
    int o = idx >> 7, c = idx & 127;
    wc[o * PITCH_K + c] = (__bf16)wconv[idx];            // Bt_conv[o][c] = w_conv[o][c]
  } else if (idx < NC + N_W1) {
    int j = idx - NC;
    int e = j / (HPAD * PITCH_K), r = j % (HPAD * PITCH_K);
    int h = r / PITCH_K, c = r % PITCH_K;
    float v = (h < HIDDEN && c < 128) ? w1[((size_t)e * 128 + c) * HIDDEN + h] : 0.f;
    wb1[j] = (__bf16)v;                                  // Bt1[e][h][c] = w1[e][c][h]
  } else if (idx < NC + N_W1 + N_W2) {
    int j = idx - NC - N_W1;
    int e = j / (128 * PITCH_H), r = j % (128 * PITCH_H);
    int o = r / PITCH_H, h = r % PITCH_H;
    float v = (h < HIDDEN) ? w2[((size_t)e * HIDDEN + h) * 128 + o] : 0.f;
    wb2[j] = (__bf16)v;                                  // Bt2[e][o][h] = w2[e][h][o]
  }
}

// ---------------- fused conv + gating + MoE main kernel ----------------
__global__ __launch_bounds__(256, 1)
void k_main(const float* __restrict__ x, const float* __restrict__ wgate,
            const float* __restrict__ b1g, const float* __restrict__ b2g,
            char* __restrict__ ws, float* __restrict__ out) {
  extern __shared__ char smem[];
  float* sX = (float*)(smem + L_SX);
  __bf16* sSF = (__bf16*)(smem + L_SSF);
  __bf16* sTY = (__bf16*)(smem + L_STY);
  __bf16* sH = (__bf16*)(smem + L_SH);
  __bf16* sW = (__bf16*)(smem + L_SW);
  float* sG = (float*)(smem + L_SG);
  float* sSc = (float*)(smem + L_SSC);

  const int t = threadIdx.x;
  const int lane = t & 31, w = t >> 5;
  const int nl = lane & 15;
  const int off8 = (lane < 16) ? 0 : 8;    // C/D row offset, A k-half
  const int off16 = off8 * 2;              // B k-half

  const int b = blockIdx.x / WGS_PER_IMG;
  const int hw0 = (blockIdx.x % WGS_PER_IMG) * TOK_PER_WG;
  const float* xbase = x + (size_t)b * 128 * HW + hw0;
  const float* ws_scale = (const float*)(ws + OFF_SCALE);
  float* ws_stats = (float*)(ws + OFF_STATS);

  // ---- stage raw x tokens [tok][c] ----
  for (int i = 0; i < 64; ++i) {
    int idx = i * 256 + t;
    int c = idx >> 7, tok = idx & 127;
    sX[tok * PITCH_X + c] = xbase[(size_t)c * HW + tok];
  }
  if (t < 128) sSc[t] = ws_scale[b * 128 + t];
  __syncthreads();

  // ---- sf = scale * x (bf16) ----
  for (int i = 0; i < 64; ++i) {
    int idx = i * 256 + t;
    int tok = idx >> 7, c = idx & 127;
    sSF[tok * PITCH_K + c] = (__bf16)(sX[tok * PITCH_X + c] * sSc[c]);
  }

  // ---- gating (fp32, exact routing) ----
  float* sWg = sG;  // alias: w_gate staged, then overwritten by gates after barrier
  sWg[t] = wgate[t];
  sWg[t + 256] = wgate[t + 256];
  __syncthreads();
  {
    int tok = t >> 1, e0 = (t & 1) * 2;
    float f0 = 0.f, f1 = 0.f;
    for (int k = 0; k < 128; ++k) {
      float xv = sX[tok * PITCH_X + k];
      f0 = fmaf(xv, sWg[k * 4 + e0], f0);
      f1 = fmaf(xv, sWg[k * 4 + e0 + 1], f1);
    }
    float p0 = __shfl_xor(f0, 1, 32), p1 = __shfl_xor(f1, 1, 32);
    float l[4];
    if ((t & 1) == 0) { l[0] = f0; l[1] = f1; l[2] = p0; l[3] = p1; }
    else { l[0] = p0; l[1] = p1; l[2] = f0; l[3] = f1; }
    int bi = 0; float bv = l[0];
#pragma unroll
    for (int j = 1; j < 4; ++j) if (l[j] > bv) { bv = l[j]; bi = j; }
    int si = 0; float sv = -3.4e38f;
#pragma unroll
    for (int j = 0; j < 4; ++j) if (j != bi && l[j] > sv) { sv = l[j]; si = j; }
    float g1 = 1.f / (1.f + expf(sv - bv));   // softmax over [top1, top2]
    float g2 = 1.f - g1;
    __syncthreads();  // all sWg reads done; safe to overwrite with gates
    if ((t & 1) == 0) {
#pragma unroll
      for (int j = 0; j < 4; ++j) sG[tok * 4 + j] = (j == bi) ? g1 : ((j == si) ? g2 : 0.f);
    }
  }
  __syncthreads();
  if (t < 8) {  // importance / load partial sums -> global fp32 atomics
    int e = t & 3; float s = 0.f;
    if (t < 4) { for (int k = 0; k < 128; ++k) s += sG[k * 4 + e]; atomicAdd(&ws_stats[e], s); }
    else { for (int k = 0; k < 128; ++k) s += (sG[k * 4 + e] > 0.f) ? 1.f : 0.f; atomicAdd(&ws_stats[4 + e], s); }
  }

  // ---- conv GEMM: ty = sf @ w_conv^T ----
  {
    const uint4* src = (const uint4*)(ws + OFF_WC);
    uint4* dst = (uint4*)sW;
    for (int i = t; i < (128 * PITCH_K * 2) / 16; i += 256) dst[i] = src[i];
  }
  __syncthreads();
  {
    const __bf16* arow = sSF + (w * 16 + nl) * PITCH_K;
    v16bf aC[4];
#pragma unroll
    for (int ks = 0; ks < 4; ++ks) aC[ks] = ldfragA(arow, ks * 32, off8);
#pragma unroll
    for (int nt = 0; nt < 8; ++nt) {
      v8f acc = {};
      const __bf16* brow = sW + (nt * 16 + nl) * PITCH_K;
#pragma unroll
      for (int ks = 0; ks < 4; ++ks)
        acc = wmma_bf16(aC[ks], ldfragB(brow, ks * 32, off16), acc);
#pragma unroll
      for (int r = 0; r < 8; ++r)
        sTY[(w * 16 + r + off8) * PITCH_K + nt * 16 + nl] = (__bf16)acc[r];
    }
  }
  __syncthreads();

  // ---- experts ----
  float yacc[8][8];
#pragma unroll
  for (int i = 0; i < 8; ++i)
#pragma unroll
    for (int j = 0; j < 8; ++j) yacc[i][j] = 0.f;

  // fc1 A-fragments (sTY is stable for the whole expert loop) -> load once
  v16bf a1[4];
  {
    const __bf16* arow = sTY + (w * 16 + nl) * PITCH_K;
#pragma unroll
    for (int ks = 0; ks < 4; ++ks) a1[ks] = ldfragA(arow, ks * 32, off8);
  }

  for (int e = 0; e < NEXP; ++e) {
    if (lane == 0) __builtin_prefetch(ws + OFF_W2 + (size_t)e * 128 * PITCH_H * 2, 0, 1);
    {  // stage fc1 Bt
      const uint4* src = (const uint4*)(ws + OFF_W1 + (size_t)e * HPAD * PITCH_K * 2);
      uint4* dst = (uint4*)sW;
      for (int i = t; i < (HPAD * PITCH_K * 2) / 16; i += 256) dst[i] = src[i];
    }
    __syncthreads();
    {  // fc1 + exact GELU -> sH
      for (int nt = 0; nt < 22; ++nt) {
        v8f acc = {};
        const __bf16* brow = sW + (nt * 16 + nl) * PITCH_K;
#pragma unroll
        for (int ks = 0; ks < 4; ++ks)
          acc = wmma_bf16(a1[ks], ldfragB(brow, ks * 32, off16), acc);
        int hc = nt * 16 + nl;
        float bv = (hc < HIDDEN) ? b1g[e * HIDDEN + hc] : 0.f;
#pragma unroll
        for (int r = 0; r < 8; ++r) {
          float f = acc[r] + bv;
          float g = 0.5f * f * (1.f + erff(f * 0.70710678118654752f));
          sH[(w * 16 + r + off8) * PITCH_H + hc] = (__bf16)g;
        }
      }
    }
    __syncthreads();
    {  // stage fc2 Bt
      const uint4* src = (const uint4*)(ws + OFF_W2 + (size_t)e * 128 * PITCH_H * 2);
      uint4* dst = (uint4*)sW;
      for (int i = t; i < (128 * PITCH_H * 2) / 16; i += 256) dst[i] = src[i];
    }
    __syncthreads();
    {  // fc2 + gated accumulate
      float g8[8];
#pragma unroll
      for (int r = 0; r < 8; ++r) g8[r] = sG[(w * 16 + r + off8) * 4 + e];
      // fc2 A-fragments (this expert's h) -> load once, reuse across 8 n-tiles
      v16bf a2[11];
      {
        const __bf16* arow = sH + (w * 16 + nl) * PITCH_H;
#pragma unroll
        for (int ks = 0; ks < 11; ++ks) a2[ks] = ldfragA(arow, ks * 32, off8);
      }
#pragma unroll
      for (int nt = 0; nt < 8; ++nt) {
        v8f acc = {};
        const __bf16* brow = sW + (nt * 16 + nl) * PITCH_H;
#pragma unroll
        for (int ks = 0; ks < 11; ++ks)
          acc = wmma_bf16(a2[ks], ldfragB(brow, ks * 32, off16), acc);
        float b2v = b2g[e * 128 + nt * 16 + nl];
#pragma unroll
        for (int r = 0; r < 8; ++r) yacc[nt][r] = fmaf(g8[r], acc[r] + b2v, yacc[nt][r]);
      }
    }
    __syncthreads();
  }

  // ---- residual add, stage y f32 (reuse sH region), coalesced store ----
  float* sY = (float*)(smem + L_SH);
#pragma unroll
  for (int nt = 0; nt < 8; ++nt)
#pragma unroll
    for (int r = 0; r < 8; ++r) {
      int tok = w * 16 + r + off8, c = nt * 16 + nl;
      sY[tok * PITCH_X + c] = yacc[nt][r] + sX[tok * PITCH_X + c];
    }
  __syncthreads();
  float* obase = out + (size_t)b * 128 * HW + hw0;
  for (int i = 0; i < 16; ++i) {
    int qi = i * 256 + t;
    int c = qi >> 5, tq = (qi & 31) * 4;
    float4 v;
    v.x = sY[(tq + 0) * PITCH_X + c];
    v.y = sY[(tq + 1) * PITCH_X + c];
    v.z = sY[(tq + 2) * PITCH_X + c];
    v.w = sY[(tq + 3) * PITCH_X + c];
    *(float4*)(obase + (size_t)c * HW + tq) = v;
  }
}

// ---------------- loss ----------------
__global__ void k_loss(const char* __restrict__ ws, float* __restrict__ out, int out_size) {
  if (threadIdx.x == 0 && blockIdx.x == 0) {
    const float* st = (const float*)(ws + OFF_STATS);
    float mi = 0.f, ml = 0.f;
    for (int e = 0; e < 4; ++e) { mi += st[e]; ml += st[4 + e]; }
    mi *= 0.25f; ml *= 0.25f;
    float vi = 0.f, vl = 0.f;
    for (int e = 0; e < 4; ++e) {
      float di = st[e] - mi, dl = st[4 + e] - ml;
      vi += di * di; vl += dl * dl;
    }
    vi *= 0.25f; vl *= 0.25f;
    out[out_size - 1] = 0.01f * (vi / (mi * mi + 1e-10f) + vl / (ml * ml + 1e-10f));
  }
}

extern "C" void kernel_launch(void* const* d_in, const int* in_sizes, int n_in,
                              void* d_out, int out_size, void* d_ws, size_t ws_size,
                              hipStream_t stream) {
  const float* x = (const float*)d_in[0];
  const float* prompt = (const float*)d_in[1];
  const float* w_lin = (const float*)d_in[2];
  const float* b_lin = (const float*)d_in[3];
  const float* w_conv = (const float*)d_in[4];
  const float* w_gate = (const float*)d_in[5];
  const float* w1 = (const float*)d_in[6];
  const float* b1 = (const float*)d_in[7];
  const float* w2 = (const float*)d_in[8];
  const float* b2 = (const float*)d_in[9];
  float* out = (float*)d_out;
  char* ws = (char*)d_ws;

  (void)in_sizes; (void)n_in; (void)ws_size;
  hipFuncSetAttribute((const void*)k_main, hipFuncAttributeMaxDynamicSharedMemorySize, SMEM_MAIN);

  k_emb<<<256, 256, 0, stream>>>(x, (float*)(ws + OFF_EMB));
  k_spectral<<<1, 64, 0, stream>>>(prompt, w_lin, b_lin, ws);
  const int prep_total = 128 * 128 + N_W1 + N_W2;
  k_prep<<<(prep_total + 255) / 256, 256, 0, stream>>>(w_conv, w1, w2, ws);
  k_main<<<NWG, 256, SMEM_MAIN, stream>>>(x, w_gate, b1, b2, ws, out);
  k_loss<<<1, 32, 0, stream>>>(ws, out, out_size);
}
